// GNNBranch_66846870995381
// MI455X (gfx1250) — matrix operand order repaired and checked by reading, not compiled, not run on hardware
//
#include <hip/hip_runtime.h>
#include <hip/hip_bf16.h>

// ---------------- problem constants ----------------
#define N_NODES 40000
#define N_EDGES 640000
#define ETOT    (N_EDGES + N_NODES)   // with self loops
#define F_IN    64
#define HID     128
#define OUT_F   64
#define HEADS   4
#define NG      64
#define NCOLS   (HEADS * HID)         // 512

typedef float        v8f  __attribute__((ext_vector_type(8)));
typedef __bf16       v16bf __attribute__((ext_vector_type(16)));

union Frag16 { v16bf v; uint4 q[2]; };

__device__ __forceinline__ unsigned short f2bf(float f) {
    unsigned int u = __float_as_uint(f);
    unsigned int r = u + 0x7FFFu + ((u >> 16) & 1u);   // round-to-nearest-even
    return (unsigned short)(r >> 16);
}

// LDS byte-offset of a generic pointer into a __shared__ object (addrspacecast -> ptrtoint)
__device__ __forceinline__ unsigned lds_offset(const void* p) {
    return (unsigned)(unsigned long long)(__attribute__((address_space(3))) const void*)p;
}

// ---------------- elementwise / fill ----------------
__global__ void k_fill_f32(float* p, float v, int n) {
    int i = blockIdx.x * blockDim.x + threadIdx.x;
    if (i < n) p[i] = v;
}
__global__ void k_fill_u32(unsigned int* p, unsigned int v, int n) {
    int i = blockIdx.x * blockDim.x + threadIdx.x;
    if (i < n) p[i] = v;
}
__global__ void k_f32_to_bf16(const float* __restrict__ in, unsigned short* __restrict__ out, int n) {
    int i = blockIdx.x * blockDim.x + threadIdx.x;
    if (i < n) out[i] = f2bf(in[i]);
}

// ---------------- WMMA GEMM: C[M,512] = A[M,K] @ B[K,512], bf16 in, f32 out -----
// block = 256 threads (8 waves). Block tile: 32 rows x 128 cols.
// wave w: rows (w/4)*16, cols (w%4)*32 (two 16x16 N-tiles -> A-frag reuse, two
// independent WMMA accumulator chains per wave to cover the bf16 RAW hazard).
// A tile is filled with GLOBAL_LOAD_ASYNC_TO_LDS_B128 (ASYNCcnt); B tile is
// transposed into LDS on the scalar path.
template<int K>
__global__ __launch_bounds__(256) void k_gemm_wmma(const unsigned short* __restrict__ A,
                                                   const unsigned short* __restrict__ B,
                                                   float* __restrict__ C) {
    constexpr int BM = 32, BN = 128;
    constexpr int LDA = K + 8;                       // +16B pad -> conflict-free b128 frag loads
    __shared__ __align__(16) unsigned short As[BM * LDA];   // [row][k]
    __shared__ __align__(16) unsigned short Bs[BN * LDA];   // transposed: [col][k]

    const int tid = threadIdx.x;
    const int m0  = blockIdx.y * BM;
    const int n0b = blockIdx.x * BN;

    // --- async fill of A tile: BM*K/8 16-byte chunks, uniform trip count ---
    constexpr int ACHUNKS = BM * K / 8;              // 16B chunks
    const unsigned as_base = lds_offset(As);
#pragma unroll
    for (int i = tid; i < ACHUNKS; i += 256) {
        int r  = i / (K / 8);
        int kc = i % (K / 8);
        unsigned long long ga = (unsigned long long)(A + (size_t)(m0 + r) * K + kc * 8);
        unsigned la = as_base + (unsigned)(r * LDA + kc * 8) * 2u;
        asm volatile("global_load_async_to_lds_b128 %0, %1, off"
                     :: "v"(la), "v"(ga) : "memory");
    }
    // --- scalar transposed fill of B tile ---
    for (int i = tid; i < BN * K; i += 256) {
        int n = i / K, k = i % K;
        Bs[n * LDA + k] = B[(size_t)k * NCOLS + n0b + n];
    }
    asm volatile("s_wait_asynccnt 0" ::: "memory");
    __syncthreads();

    const int wave = tid >> 5;
    const int lane = tid & 31;
    const int mrow = (wave >> 2) * 16;
    const int ncol = (wave & 3) * 32;                // two 16-wide N tiles
    const int lmod = lane & 15;
    const int hi   = lane >> 4;

    v8f acc0 = {}, acc1 = {};
#pragma unroll
    for (int kb = 0; kb < K; kb += 32) {
        // A 16x32 bf16 frag: lanes<16 hold K {kb..kb+7, kb+16..kb+23}; lanes>=16 shift by 8
        const unsigned short* ap = &As[(mrow + lmod) * LDA + kb + hi * 8];
        Frag16 a;
        a.q[0] = *(const uint4*)ap;
        a.q[1] = *(const uint4*)(ap + 16);
        // B 32x16 bf16 frags: lanes<16 hold K {kb..kb+7, kb+16..kb+23}; lanes>=16 shift by 8
        const unsigned short* bp0 = &Bs[(ncol + lmod) * LDA + kb + hi * 8];
        const unsigned short* bp1 = bp0 + 16 * LDA;
        Frag16 b0, b1;
        b0.q[0] = *(const uint4*)bp0;
        b0.q[1] = *(const uint4*)(bp0 + 16);
        b1.q[0] = *(const uint4*)bp1;
        b1.q[1] = *(const uint4*)(bp1 + 16);
        acc0 = __builtin_amdgcn_wmma_f32_16x16x32_bf16(false, a.v, false, b0.v,
                                                       (short)0, acc0, false, false);
        acc1 = __builtin_amdgcn_wmma_f32_16x16x32_bf16(false, a.v, false, b1.v,
                                                       (short)0, acc1, false, false);
    }

    // C/D layout: VGPR i -> row m0+mrow + i + hi*8, col n0b+ncol+lmod
    const int gm = m0 + mrow + hi * 8;
    const int gn = n0b + ncol + lmod;
#pragma unroll
    for (int i = 0; i < 8; ++i) {
        C[(size_t)(gm + i) * NCOLS + gn]      = acc0[i];
        C[(size_t)(gm + i) * NCOLS + gn + 16] = acc1[i];
    }
}

// ---------------- attention scores: s[n,h] = <xh[n,h,:], a[h,:]> ----------------
__global__ __launch_bounds__(128) void k_scores(const float* __restrict__ xh,
                                                const float* __restrict__ a_s,
                                                const float* __restrict__ a_d,
                                                float* __restrict__ ssrc,
                                                float* __restrict__ sdst) {
    const int n = blockIdx.x;
    const int h = threadIdx.x >> 5;
    const int lane = threadIdx.x & 31;
    const float* row = xh + (size_t)n * NCOLS + h * HID;
    float s1 = 0.f, s2 = 0.f;
    for (int c = lane; c < HID; c += 32) {
        float v = row[c];
        s1 += v * a_s[h * HID + c];
        s2 += v * a_d[h * HID + c];
    }
    for (int off = 16; off > 0; off >>= 1) {
        s1 += __shfl_down(s1, off, 32);
        s2 += __shfl_down(s2, off, 32);
    }
    if (lane == 0) { ssrc[n * HEADS + h] = s1; sdst[n * HEADS + h] = s2; }
}

// ---------------- edge pass 1: leaky-relu score + segment max (ordered-uint) ----
__global__ void k_edge_max(const int* __restrict__ ei,
                           const float* __restrict__ ssrc, const float* __restrict__ sdst,
                           float* __restrict__ ebuf, unsigned int* __restrict__ mbuf) {
    int e = blockIdx.x * blockDim.x + threadIdx.x;
    if (e >= ETOT) return;
    int s = (e < N_EDGES) ? ei[e]           : (e - N_EDGES);
    int d = (e < N_EDGES) ? ei[N_EDGES + e] : (e - N_EDGES);
#pragma unroll
    for (int h = 0; h < HEADS; ++h) {
        float v = ssrc[s * HEADS + h] + sdst[d * HEADS + h];
        v = (v > 0.f) ? v : 0.2f * v;                 // leaky_relu(0.2)
        ebuf[(size_t)e * HEADS + h] = v;
        unsigned int u = __float_as_uint(v);
        u = (u & 0x80000000u) ? ~u : (u | 0x80000000u);
        atomicMax(&mbuf[d * HEADS + h], u);
    }
}

// ---------------- edge pass 2: p = exp(e - m[dst]); denom += p -------------------
__global__ void k_edge_exp(const int* __restrict__ ei,
                           float* __restrict__ ebuf, const unsigned int* __restrict__ mbuf,
                           float* __restrict__ denom) {
    int e = blockIdx.x * blockDim.x + threadIdx.x;
    if (e >= ETOT) return;
    int d = (e < N_EDGES) ? ei[N_EDGES + e] : (e - N_EDGES);
#pragma unroll
    for (int h = 0; h < HEADS; ++h) {
        unsigned int u = mbuf[d * HEADS + h];
        float m = (u & 0x80000000u) ? __uint_as_float(u ^ 0x80000000u) : __uint_as_float(~u);
        float p = __expf(ebuf[(size_t)e * HEADS + h] - m);
        ebuf[(size_t)e * HEADS + h] = p;
        atomicAdd(&denom[d * HEADS + h], p);
    }
}

// ---------------- edge pass 3: agg[dst,c] += 0.25 * sum_h xh[src,h,c]*alpha -----
// one thread per (edge, 4-channel group) -> b128 gathers of L2-resident xh
__global__ __launch_bounds__(256) void k_edge_msg(const int* __restrict__ ei,
                                                  const float* __restrict__ xh,
                                                  const float* __restrict__ ebuf,
                                                  const float* __restrict__ denom,
                                                  float* __restrict__ agg) {
    int idx = blockIdx.x * blockDim.x + threadIdx.x;     // ETOT*32 threads
    int e  = idx >> 5;
    int c4 = (idx & 31) << 2;                            // channel group of 4
    if (e >= ETOT) return;
    int s = (e < N_EDGES) ? ei[e]           : (e - N_EDGES);
    int d = (e < N_EDGES) ? ei[N_EDGES + e] : (e - N_EDGES);
    const float* xr = xh + (size_t)s * NCOLS + c4;
    __builtin_prefetch(xr, 0, 0);                        // global_prefetch_b8 (L2-resident xh)
    float4 acc = make_float4(0.f, 0.f, 0.f, 0.f);
#pragma unroll
    for (int h = 0; h < HEADS; ++h) {
        float alpha = ebuf[(size_t)e * HEADS + h] / denom[d * HEADS + h];
        float4 v = *(const float4*)(xr + h * HID);       // global_load_b128
        acc.x += v.x * alpha; acc.y += v.y * alpha;
        acc.z += v.z * alpha; acc.w += v.w * alpha;
    }
    float* ag = agg + (size_t)d * HID + c4;
    atomicAdd(ag + 0, 0.25f * acc.x);                    // head-mean folded in
    atomicAdd(ag + 1, 0.25f * acc.y);
    atomicAdd(ag + 2, 0.25f * acc.z);
    atomicAdd(ag + 3, 0.25f * acc.w);
}

// ---------------- node epilogue: +bias, BN(eval), ReLU; emit f32 + bf16 ---------
__global__ void k_node_post(const float* __restrict__ agg,
                            const float* __restrict__ b, const float* __restrict__ g,
                            const float* __restrict__ be,
                            float* __restrict__ hf, unsigned short* __restrict__ hb) {
    int idx = blockIdx.x * blockDim.x + threadIdx.x;     // N*HID
    int c = idx & 127;
    float v = agg[idx] + b[c];
    v = v * (g[c] * rsqrtf(1.0f + 1e-5f)) + be[c];
    v = fmaxf(v, 0.f);
    hf[idx] = v;
    hb[idx] = f2bf(v);
}

// ---------------- global mean pool -----------------------------------------------
__global__ void k_pool(const float* __restrict__ h, const int* __restrict__ batch,
                       float* __restrict__ sums, float* __restrict__ cnt) {
    int idx = blockIdx.x * blockDim.x + threadIdx.x;     // N*HID
    int n = idx >> 7;
    int c = idx & 127;
    int bg = batch[n];
    atomicAdd(&sums[bg * HID + c], h[idx]);
    if (c == 0) atomicAdd(&cnt[bg], 1.0f);
}

// ---------------- FC head: pooled -> relu(fc1) -> fc2 ----------------------------
__global__ __launch_bounds__(128) void k_head(const float* __restrict__ sums,
                                              const float* __restrict__ cnt,
                                              const float* __restrict__ w1, const float* __restrict__ b1,
                                              const float* __restrict__ w2, const float* __restrict__ b2,
                                              float* __restrict__ out) {
    __shared__ float pooled[HID];
    __shared__ float hid[HID];
    int g = blockIdx.x;
    int t = threadIdx.x;
    float cc = fmaxf(cnt[g], 1.0f);
    pooled[t] = sums[g * HID + t] / cc;
    __syncthreads();
    float a = b1[t];
    for (int k = 0; k < HID; ++k) a += pooled[k] * w1[k * HID + t];
    hid[t] = fmaxf(a, 0.f);
    __syncthreads();
    if (t < OUT_F) {
        float o = b2[t];
        for (int j = 0; j < HID; ++j) o += hid[j] * w2[j * OUT_F + t];
        out[g * OUT_F + t] = o;
    }
}

// =================================================================================
extern "C" void kernel_launch(void* const* d_in, const int* in_sizes, int n_in,
                              void* d_out, int out_size, void* d_ws, size_t ws_size,
                              hipStream_t stream) {
    const float* x     = (const float*)d_in[0];
    const int*   ei    = (const int*)d_in[1];       // edge_index [2,E]
    const int*   batch = (const int*)d_in[2];
    const float* W[3]  = { (const float*)d_in[3],  (const float*)d_in[9],  (const float*)d_in[15] };
    const float* AS[3] = { (const float*)d_in[4],  (const float*)d_in[10], (const float*)d_in[16] };
    const float* AD[3] = { (const float*)d_in[5],  (const float*)d_in[11], (const float*)d_in[17] };
    const float* Bb[3] = { (const float*)d_in[6],  (const float*)d_in[12], (const float*)d_in[18] };
    const float* Gg[3] = { (const float*)d_in[7],  (const float*)d_in[13], (const float*)d_in[19] };
    const float* Be[3] = { (const float*)d_in[8],  (const float*)d_in[14], (const float*)d_in[20] };
    const float* fc1w  = (const float*)d_in[21];
    const float* fc1b  = (const float*)d_in[22];
    const float* fc2w  = (const float*)d_in[23];
    const float* fc2b  = (const float*)d_in[24];
    float* out = (float*)d_out;

    // ---- workspace carve-up (256B aligned) ----
    char* base = (char*)d_ws;
    size_t off = 0;
    auto carve = [&](size_t bytes) { size_t o = off; off = (off + bytes + 255) & ~(size_t)255; return o; };
    unsigned short* wb0 = (unsigned short*)(base + carve((size_t)F_IN * NCOLS * 2));
    unsigned short* wb1 = (unsigned short*)(base + carve((size_t)HID  * NCOLS * 2));
    unsigned short* wb2 = (unsigned short*)(base + carve((size_t)HID  * NCOLS * 2));
    unsigned short* hb  = (unsigned short*)(base + carve((size_t)N_NODES * HID * 2));
    float* xh    = (float*)(base + carve((size_t)N_NODES * NCOLS * 4));
    float* ssrc  = (float*)(base + carve((size_t)N_NODES * HEADS * 4));
    float* sdst  = (float*)(base + carve((size_t)N_NODES * HEADS * 4));
    float* ebuf  = (float*)(base + carve((size_t)ETOT * HEADS * 4));
    unsigned int* mbuf = (unsigned int*)(base + carve((size_t)N_NODES * HEADS * 4));
    float* denom = (float*)(base + carve((size_t)N_NODES * HEADS * 4));
    float* agg   = (float*)(base + carve((size_t)N_NODES * HID * 4));
    float* hf    = (float*)(base + carve((size_t)N_NODES * HID * 4));
    float* psum  = (float*)(base + carve((size_t)NG * HID * 4));
    float* pcnt  = (float*)(base + carve((size_t)NG * 4));
    (void)ws_size; (void)n_in; (void)in_sizes; (void)out_size;

    const unsigned int NEG_INF_ORD = 0x007FFFFFu;   // ordered-uint encoding of -inf

    // weights -> bf16 (once per launch, deterministic)
    k_f32_to_bf16<<<(F_IN * NCOLS + 255) / 256, 256, 0, stream>>>(W[0], wb0, F_IN * NCOLS);
    k_f32_to_bf16<<<(HID  * NCOLS + 255) / 256, 256, 0, stream>>>(W[1], wb1, HID * NCOLS);
    k_f32_to_bf16<<<(HID  * NCOLS + 255) / 256, 256, 0, stream>>>(W[2], wb2, HID * NCOLS);
    // x -> bf16
    k_f32_to_bf16<<<(N_NODES * F_IN) / 256, 256, 0, stream>>>(x, hb, N_NODES * F_IN);

    unsigned short* wbs[3] = { wb0, wb1, wb2 };
    const int eb   = (ETOT + 255) / 256;
    const int msg  = (ETOT * 32) / 256;      // 85000 blocks, thread = (edge, 4-chan group)
    const int nhid = (N_NODES * HID) / 256;

    for (int l = 0; l < 3; ++l) {
        // feature transform via WMMA (block tile 32x128, wave tile 16x32)
        if (l == 0)
            k_gemm_wmma<F_IN><<<dim3(NCOLS / 128, N_NODES / 32), 256, 0, stream>>>(hb, wbs[0], xh);
        else
            k_gemm_wmma<HID><<<dim3(NCOLS / 128, N_NODES / 32), 256, 0, stream>>>(hb, wbs[l], xh);

        k_fill_u32<<<(N_NODES * HEADS + 255) / 256, 256, 0, stream>>>(mbuf, NEG_INF_ORD, N_NODES * HEADS);
        k_fill_f32<<<(N_NODES * HEADS + 255) / 256, 256, 0, stream>>>(denom, 0.f, N_NODES * HEADS);
        k_fill_f32<<<nhid, 256, 0, stream>>>(agg, 0.f, N_NODES * HID);

        k_scores<<<N_NODES, 128, 0, stream>>>(xh, AS[l], AD[l], ssrc, sdst);
        k_edge_max<<<eb, 256, 0, stream>>>(ei, ssrc, sdst, ebuf, mbuf);
        k_edge_exp<<<eb, 256, 0, stream>>>(ei, ebuf, mbuf, denom);
        k_edge_msg<<<msg, 256, 0, stream>>>(ei, xh, ebuf, denom, agg);
        k_node_post<<<nhid, 256, 0, stream>>>(agg, Bb[l], Gg[l], Be[l], hf, hb);
    }

    k_fill_f32<<<(NG * HID + 255) / 256, 256, 0, stream>>>(psum, 0.f, NG * HID);
    k_fill_f32<<<1, 256, 0, stream>>>(pcnt, 0.f, NG);
    k_pool<<<nhid, 256, 0, stream>>>(hf, batch, psum, pcnt);
    k_head<<<NG, 128, 0, stream>>>(psum, pcnt, fc1w, fc1b, fc2w, fc2b, out);
}
